// Pointnet2_78469052498286
// MI455X (gfx1250) — compile-verified, matrix operand order checked by reading, not run on hardware
//
#include <hip/hip_runtime.h>
#include <hip/hip_bf16.h>

// ---------------------------------------------------------------------------
// Types for WMMA
// ---------------------------------------------------------------------------
typedef __bf16 bf16;
typedef bf16  bf16x16 __attribute__((ext_vector_type(16)));
typedef float floatx8 __attribute__((ext_vector_type(8)));

// ---------------------------------------------------------------------------
// FPS: one workgroup (1024 threads) per batch. Running min-distances live in
// registers (each thread owns indices i = tid + j*1024). Argmax via LDS tree.
// ---------------------------------------------------------------------------
template <int ITEMS>
__global__ void __launch_bounds__(1024)
fps_kernel(const float* __restrict__ xyz, int N, int npoint, int* __restrict__ out)
{
    __shared__ float rv[1024];
    __shared__ int   ri[1024];
    const int b = blockIdx.x;
    const int t = threadIdx.x;
    const float* p = xyz + (size_t)b * N * 3;
    int* o = out + (size_t)b * npoint;

    float d[ITEMS], px[ITEMS], py[ITEMS], pz[ITEMS];
#pragma unroll
    for (int j = 0; j < ITEMS; ++j) {
        d[j] = 1e10f;
        int i = t + j * 1024;
        if (i < N) { px[j] = p[i*3]; py[j] = p[i*3+1]; pz[j] = p[i*3+2]; }
        else       { px[j] = 0.f; py[j] = 0.f; pz[j] = 0.f; }
    }
    if (t == 0) o[0] = 0;
    int last = 0;
    for (int s = 1; s < npoint; ++s) {
        float cx = p[last*3], cy = p[last*3+1], cz = p[last*3+2];
        float best = -1.f; int bi = 0x7fffffff;
#pragma unroll
        for (int j = 0; j < ITEMS; ++j) {
            int i = t + j * 1024;
            if (i < N) {
                float dx = px[j]-cx, dy = py[j]-cy, dz = pz[j]-cz;
                float dd = dx*dx + dy*dy + dz*dz;
                dd = fminf(d[j], dd);
                d[j] = dd;
                if (dd > best) { best = dd; bi = i; }
            }
        }
        rv[t] = best; ri[t] = bi;
        __syncthreads();
        for (int off = 512; off > 0; off >>= 1) {
            if (t < off) {
                float ov = rv[t+off]; int oi = ri[t+off];
                if (ov > rv[t] || (ov == rv[t] && oi < ri[t])) { rv[t] = ov; ri[t] = oi; }
            }
            __syncthreads();
        }
        last = ri[0];
        if (t == 0) o[s] = last;
        __syncthreads();
    }
}

// ---------------------------------------------------------------------------
// Gather sampled centroid coordinates
// ---------------------------------------------------------------------------
__global__ void __launch_bounds__(256)
gather_xyz_kernel(const float* __restrict__ xyz, const int* __restrict__ idx,
                  float* __restrict__ out, int Nsrc, int np, int B)
{
    int g = blockIdx.x * blockDim.x + threadIdx.x;
    if (g >= B * np) return;
    int b = g / np;
    int i = idx[g];
    const float* s = xyz + ((size_t)b * Nsrc + i) * 3;
    float* o = out + (size_t)g * 3;
    o[0] = s[0]; o[1] = s[1]; o[2] = s[2];
}

// ---------------------------------------------------------------------------
// Ball query: sequential ascending-index scan == reference sort semantics
// ---------------------------------------------------------------------------
__global__ void __launch_bounds__(256)
ball_query_kernel(const float* __restrict__ xyz, const float* __restrict__ nxyz,
                  int* __restrict__ out, int Nsrc, int np, int ns, float r2, int B)
{
    int g = blockIdx.x * blockDim.x + threadIdx.x;
    if (g >= B * np) return;
    int b = g / np;
    const float* c = nxyz + (size_t)g * 3;
    float cx = c[0], cy = c[1], cz = c[2];
    const float* p = xyz + (size_t)b * Nsrc * 3;
    int* o = out + (size_t)g * ns;
    int cnt = 0, first = -1;
    for (int i = 0; i < Nsrc && cnt < ns; ++i) {
        float dx = p[i*3]-cx, dy = p[i*3+1]-cy, dz = p[i*3+2]-cz;
        if (dx*dx + dy*dy + dz*dz <= r2) {
            if (first < 0) first = i;
            o[cnt++] = i;
        }
    }
    if (first < 0) first = Nsrc - 1;
    for (int k = cnt; k < ns; ++k) o[k] = first;
}

// ---------------------------------------------------------------------------
// Build grouped MLP input rows: [xyz[idx]-center | feats[idx] | zero pad]
// ---------------------------------------------------------------------------
__global__ void __launch_bounds__(256)
group_build_kernel(const float* __restrict__ xyz, const float* __restrict__ nxyz,
                   const int* __restrict__ bidx, const bf16* __restrict__ feats,
                   int Cf, int Nsrc, int np, int ns, int Kp, bf16* __restrict__ X, int B)
{
    int row = blockIdx.x * blockDim.x + threadIdx.x;
    int total = B * np * ns;
    if (row >= total) return;
    int s = row % ns;
    int g = row / ns;
    int b = g / np;
    int i = bidx[(size_t)g * ns + s];
    const float* pt = xyz + ((size_t)b * Nsrc + i) * 3;
    const float* c  = nxyz + (size_t)g * 3;
    bf16* x = X + (size_t)row * Kp;
    x[0] = (bf16)(pt[0] - c[0]);
    x[1] = (bf16)(pt[1] - c[1]);
    x[2] = (bf16)(pt[2] - c[2]);
    int k = 3;
    if (feats) {
        const bf16* f = feats + ((size_t)b * Nsrc + i) * Cf;
        for (int ch = 0; ch < Cf; ++ch) x[k++] = f[ch];
    }
    for (; k < Kp; ++k) x[k] = (bf16)0.0f;
}

// ---------------------------------------------------------------------------
// Pack f32 weight (K x N row-major) into WMMA B-fragment layout, bf16,
// zero-padded to Kp x N. Fragment (kt,nt): 32 lanes x 16 bf16 (32 B / lane).
// Lane l holds column n = nt*16+(l&15); k-locals {o..o+7, o+16..o+23}, o=(l>>4)*8.
// ---------------------------------------------------------------------------
__global__ void __launch_bounds__(256)
pack_weight_kernel(const float* __restrict__ W, unsigned* __restrict__ out,
                   int K, int N, int Kp)
{
    int gid = blockIdx.x * blockDim.x + threadIdx.x;
    int NT = N >> 4, KT = Kp >> 5;
    int total = KT * NT * 32;
    if (gid >= total) return;
    int lane = gid & 31;
    int frag = gid >> 5;
    int nt = frag % NT;
    int kt = frag / NT;
    int n = (nt << 4) + (lane & 15);
    int o = (lane >> 4) << 3;
    unsigned* dst = out + (size_t)gid * 8;
    for (int j2 = 0; j2 < 8; ++j2) {
        unsigned pack = 0;
        for (int h = 0; h < 2; ++h) {
            int j  = j2 * 2 + h;
            int kl = o + j + (j >= 8 ? 8 : 0);
            int k  = (kt << 5) + kl;
            float f = (k < K) ? W[(size_t)k * N + n] : 0.f;
            union { float f; unsigned u; } cv; cv.f = f;
            unsigned r = (cv.u + 0x7FFFu + ((cv.u >> 16) & 1u)) >> 16;   // RNE f32->bf16
            pack |= (r & 0xFFFFu) << (16 * h);
        }
        dst[j2] = pack;
    }
}

// ---------------------------------------------------------------------------
// WMMA GEMM: Y = relu(X @ W + b). X: M x Kp bf16 row-major (Kp % 32 == 0),
// Wp: packed fragments, bias: f32[N]. One 16x16 output tile per wave,
// v_wmma_f32_16x16x32_bf16 over Kp/32 steps. Optional f32 output.
// ---------------------------------------------------------------------------
__global__ void __launch_bounds__(256)
gemm_bias_relu_kernel(const bf16* __restrict__ X, const uint4* __restrict__ Wp,
                      const float* __restrict__ bias, bf16* __restrict__ Ybf,
                      float* __restrict__ Yf, int M, int Kp, int Nout)
{
    const int lane  = threadIdx.x & 31;
    const int wib   = threadIdx.x >> 5;
    const int tilesN = Nout >> 4;
    const int tilesM = M >> 4;
    int tile = blockIdx.x * 8 + wib;
    if (tile >= tilesM * tilesN) return;
    const int tn = tile % tilesN;
    const int tm = tile / tilesN;
    const int row  = (tm << 4) + (lane & 15);
    const int koff = (lane >> 4) << 3;          // 0 or 8
    const int KT = Kp >> 5;

    floatx8 acc = {0.f, 0.f, 0.f, 0.f, 0.f, 0.f, 0.f, 0.f};
    const bf16* xrow = X + (size_t)row * Kp + koff;
    for (int kt = 0; kt < KT; ++kt) {
        union { bf16x16 v; uint4 u[2]; } A, Bm;
        A.u[0] = *(const uint4*)(xrow + kt * 32);
        A.u[1] = *(const uint4*)(xrow + kt * 32 + 16);
        const uint4* wp = Wp + ((((size_t)kt * tilesN + tn) << 5) + lane) * 2;
        Bm.u[0] = wp[0];
        Bm.u[1] = wp[1];
        acc = __builtin_amdgcn_wmma_f32_16x16x32_bf16(
            false, A.v, false, Bm.v, (short)0, acc, false, false);
    }
    const int n  = (tn << 4) + (lane & 15);
    const float bn = bias[n];
    const int mtop = (tm << 4) + ((lane >> 4) << 3);
#pragma unroll
    for (int r = 0; r < 8; ++r) {
        float v = acc[r] + bn;
        v = v > 0.f ? v : 0.f;
        size_t off = (size_t)(mtop + r) * Nout + n;
        Ybf[off] = (bf16)v;
        if (Yf) Yf[off] = v;
    }
}

// ---------------------------------------------------------------------------
// Max-pool over nsample; writes f32 (for FP interpolation) and bf16 (for next
// stage group gather).
// ---------------------------------------------------------------------------
__global__ void __launch_bounds__(256)
maxpool_kernel(const bf16* __restrict__ Y, float* __restrict__ outf,
               bf16* __restrict__ outb, int np, int ns, int C, int B)
{
    int g = blockIdx.x * blockDim.x + threadIdx.x;
    int total = B * np * C;
    if (g >= total) return;
    int c = g % C;
    int q = g / C;
    const bf16* y = Y + ((size_t)q * ns) * C + c;
    float m = -1e30f;
    for (int s = 0; s < ns; ++s) {
        float v = (float)y[(size_t)s * C];
        m = v > m ? v : m;
    }
    outf[g] = m;
    outb[g] = (bf16)m;
}

// ---------------------------------------------------------------------------
// FP stage: 3-NN inverse-distance interpolation + concat -> bf16 GEMM input
// ---------------------------------------------------------------------------
__global__ void __launch_bounds__(256)
fp_interp_kernel(const float* __restrict__ uxyz, const float* __restrict__ kxyz,
                 const float* __restrict__ ufeat, const float* __restrict__ kfeat,
                 int nU, int nK, int Cu, int Ck, int Kp, bf16* __restrict__ X, int B)
{
    int g = blockIdx.x * blockDim.x + threadIdx.x;
    if (g >= B * nU) return;
    int b = g / nU;
    const float* u = uxyz + (size_t)g * 3;
    const float* kp = kxyz + (size_t)b * nK * 3;
    float ux = u[0], uy = u[1], uz = u[2];
    float d0 = 1e30f, d1 = 1e30f, d2 = 1e30f;
    int i0 = 0, i1 = 0, i2 = 0;
    for (int i = 0; i < nK; ++i) {
        float dx = kp[i*3]-ux, dy = kp[i*3+1]-uy, dz = kp[i*3+2]-uz;
        float dd = dx*dx + dy*dy + dz*dz;
        if (dd < d0)      { d2 = d1; i2 = i1; d1 = d0; i1 = i0; d0 = dd; i0 = i; }
        else if (dd < d1) { d2 = d1; i2 = i1; d1 = dd; i1 = i; }
        else if (dd < d2) { d2 = dd; i2 = i; }
    }
    float w0 = 1.f/(d0 + 1e-8f), w1 = 1.f/(d1 + 1e-8f), w2 = 1.f/(d2 + 1e-8f);
    float ws = w0 + w1 + w2;
    w0 /= ws; w1 /= ws; w2 /= ws;
    const float* f0 = kfeat + ((size_t)b * nK + i0) * Ck;
    const float* f1 = kfeat + ((size_t)b * nK + i1) * Ck;
    const float* f2 = kfeat + ((size_t)b * nK + i2) * Ck;
    bf16* x = X + (size_t)g * Kp;
    for (int c = 0; c < Ck; ++c)
        x[c] = (bf16)(w0 * f0[c] + w1 * f1[c] + w2 * f2[c]);
    const float* uf = ufeat + (size_t)g * Cu;
    for (int c = 0; c < Cu; ++c)
        x[Ck + c] = (bf16)uf[c];
}

// ---------------------------------------------------------------------------
// Host orchestration
// ---------------------------------------------------------------------------
static inline unsigned cdiv(size_t a, unsigned b) { return (unsigned)((a + b - 1) / b); }

extern "C" void kernel_launch(void* const* d_in, const int* in_sizes, int n_in,
                              void* d_out, int out_size, void* d_ws, size_t ws_size,
                              hipStream_t stream)
{
    (void)in_sizes; (void)n_in; (void)out_size; (void)ws_size;
    const int B = 4, N0 = 20000;
    const int   np_[4] = {2048, 1024, 512, 256};
    const int   ns_[4] = {64, 32, 32, 16};
    const float r2_[4] = {0.01f, 0.04f, 0.16f, 0.64f};
    const int  Cout_[4] = {64, 128, 256, 256};
    const int  Kp0_[4]  = {32, 96, 160, 288};
    const int  gK_[4][3]  = {{3,32,32},{67,64,64},{131,128,128},{259,128,128}};
    const int  gN_[4][3]  = {{32,32,64},{64,64,128},{128,128,256},{128,128,256}};
    const int  gKp_[4][3] = {{32,32,32},{96,64,64},{160,128,128},{288,128,128}};
    const int  fK_[2][2] = {{512,256},{384,128}};
    const int  fN_[2][2] = {{256,256},{128,256}};

    // ---- input pointers (tree-flattened dict order) ----
    const float* pc = (const float*)d_in[0];
    const float* saw[4][3]; const float* sab[4][3];
    int base = 1;
    for (int l = 0; l < 4; ++l) {
        for (int j = 0; j < 3; ++j) saw[l][j] = (const float*)d_in[base + j];
        for (int j = 0; j < 3; ++j) sab[l][j] = (const float*)d_in[base + 3 + j];
        base += 6;
    }
    const float* fpw[2][2]; const float* fpb[2][2];
    for (int l = 0; l < 2; ++l) {
        for (int j = 0; j < 2; ++j) fpw[l][j] = (const float*)d_in[base + j];
        for (int j = 0; j < 2; ++j) fpb[l][j] = (const float*)d_in[base + 2 + j];
        base += 4;
    }

    // ---- scratch layout ----
    char* ws = (char*)d_ws;
    size_t off = 0;
    auto alloc = [&](size_t bytes) -> char* {
        char* p = ws + off;
        off += (bytes + 255) & ~(size_t)255;
        return p;
    };
    int*   fpsidx[4]; float* nxyz[4]; int* bidx[4]; float* poolf[4]; bf16* poolb[4];
    for (int l = 0; l < 4; ++l) {
        fpsidx[l] = (int*)  alloc((size_t)B * np_[l] * 4);
        nxyz[l]   = (float*)alloc((size_t)B * np_[l] * 12);
        bidx[l]   = (int*)  alloc((size_t)B * np_[l] * ns_[l] * 4);
        poolf[l]  = (float*)alloc((size_t)B * np_[l] * Cout_[l] * 4);
        poolb[l]  = (bf16*) alloc((size_t)B * np_[l] * Cout_[l] * 2);
    }
    unsigned* wp_sa[4][3];
    for (int l = 0; l < 4; ++l)
        for (int j = 0; j < 3; ++j)
            wp_sa[l][j] = (unsigned*)alloc((size_t)gKp_[l][j] * gN_[l][j] * 2);
    unsigned* wp_fp[2][2];
    for (int l = 0; l < 2; ++l)
        for (int j = 0; j < 2; ++j)
            wp_fp[l][j] = (unsigned*)alloc((size_t)fK_[l][j] * fN_[l][j] * 2);
    float* fp1f = (float*)alloc((size_t)B * 512 * 256 * 4);
    bf16*  actA = (bf16*)alloc((size_t)524288 * 32 * 2);   // 32 MB
    bf16*  actB = (bf16*)alloc((size_t)524288 * 64 * 2);   // 64 MB

    // ---- pack all weights into WMMA B-fragment layout ----
    for (int l = 0; l < 4; ++l)
        for (int j = 0; j < 3; ++j) {
            int tot = (gKp_[l][j] / 32) * (gN_[l][j] / 16) * 32;
            pack_weight_kernel<<<cdiv(tot, 256), 256, 0, stream>>>(
                saw[l][j], wp_sa[l][j], gK_[l][j], gN_[l][j], gKp_[l][j]);
        }
    for (int l = 0; l < 2; ++l)
        for (int j = 0; j < 2; ++j) {
            int tot = (fK_[l][j] / 32) * (fN_[l][j] / 16) * 32;
            pack_weight_kernel<<<cdiv(tot, 256), 256, 0, stream>>>(
                fpw[l][j], wp_fp[l][j], fK_[l][j], fN_[l][j], fK_[l][j]);
        }

    auto gemm = [&](const bf16* X, const unsigned* Wp, const float* bias,
                    bf16* Ybf, float* Yf, int M, int Kp, int N) {
        size_t tiles = (size_t)(M / 16) * (N / 16);
        gemm_bias_relu_kernel<<<cdiv(tiles, 8), 256, 0, stream>>>(
            X, (const uint4*)Wp, bias, Ybf, Yf, M, Kp, N);
    };

    // ---- SA stages ----
    const float* xyz_prev = pc;
    int Nprev = N0;
    const bf16* fprev = nullptr;
    int Cf = 0;
    for (int l = 0; l < 4; ++l) {
        int np = np_[l], ns = ns_[l];
        switch (l) {
            case 0:  fps_kernel<20><<<B, 1024, 0, stream>>>(xyz_prev, Nprev, np, fpsidx[l]); break;
            case 1:  fps_kernel<2><<<B, 1024, 0, stream>>>(xyz_prev, Nprev, np, fpsidx[l]);  break;
            default: fps_kernel<1><<<B, 1024, 0, stream>>>(xyz_prev, Nprev, np, fpsidx[l]);  break;
        }
        gather_xyz_kernel<<<cdiv((size_t)B * np, 256), 256, 0, stream>>>(
            xyz_prev, fpsidx[l], nxyz[l], Nprev, np, B);
        ball_query_kernel<<<cdiv((size_t)B * np, 256), 256, 0, stream>>>(
            xyz_prev, nxyz[l], bidx[l], Nprev, np, ns, r2_[l], B);
        int M = B * np * ns;
        group_build_kernel<<<cdiv((size_t)M, 256), 256, 0, stream>>>(
            xyz_prev, nxyz[l], bidx[l], fprev, Cf, Nprev, np, ns, Kp0_[l], actA, B);
        gemm(actA, wp_sa[l][0], sab[l][0], actB, nullptr, M, gKp_[l][0], gN_[l][0]);
        gemm(actB, wp_sa[l][1], sab[l][1], actA, nullptr, M, gKp_[l][1], gN_[l][1]);
        gemm(actA, wp_sa[l][2], sab[l][2], actB, nullptr, M, gKp_[l][2], gN_[l][2]);
        maxpool_kernel<<<cdiv((size_t)B * np * Cout_[l], 256), 256, 0, stream>>>(
            actB, poolf[l], poolb[l], np, ns, Cout_[l], B);
        xyz_prev = nxyz[l];
        Nprev = np;
        fprev = poolb[l];
        Cf = Cout_[l];
    }

    // ---- FP1: unknown = level2 (512 pts), known = level3 (256 pts) ----
    fp_interp_kernel<<<cdiv((size_t)B * 512, 256), 256, 0, stream>>>(
        nxyz[2], nxyz[3], poolf[2], poolf[3], 512, 256, 256, 256, 512, actA, B);
    {
        int M = B * 512;
        gemm(actA, wp_fp[0][0], fpb[0][0], actB, nullptr, M, 512, 256);
        gemm(actB, wp_fp[0][1], fpb[0][1], actA, fp1f,    M, 256, 256);
    }

    // ---- FP2: unknown = level1 (1024 pts), known = level2 (512 pts) ----
    fp_interp_kernel<<<cdiv((size_t)B * 1024, 256), 256, 0, stream>>>(
        nxyz[1], nxyz[2], poolf[1], fp1f, 1024, 512, 128, 256, 384, actA, B);
    {
        int M = B * 1024;
        gemm(actA, wp_fp[1][0], fpb[1][0], actB, nullptr,        M, 384, 128);
        gemm(actB, wp_fp[1][1], fpb[1][1], actA, (float*)d_out,  M, 128, 256);
    }
}
